// encoder_38293928411466
// MI455X (gfx1250) — matrix operand order repaired and checked by reading, not compiled, not run on hardware
//
#include <hip/hip_runtime.h>
#include <hip/hip_bf16.h>

typedef __bf16 bf16;
typedef __attribute__((ext_vector_type(16))) __bf16 v16bf;
typedef __attribute__((ext_vector_type(8)))  __bf16 v8bf;
typedef __attribute__((ext_vector_type(8)))  float  v8f;
typedef int v4i __attribute__((vector_size(16)));
typedef __attribute__((address_space(1))) v4i* as1_v4i;
typedef __attribute__((address_space(3))) v4i* as3_v4i;

#define SLEN  512
#define BB    16
#define HDIM  768
#define NHEAD 12
#define HD    64
#define H5    3840
#define H3    2304
#define KSTEP 64

// ---------------------------------------------------------------------------
// CDNA5 async global->LDS copy (ASYNCcnt path), guarded so the file compiles
// on toolchains without the builtin (falls back to synchronous vector copy).
// Builtin signature (from hipcc diagnostic): param0 is int4* in the global
// (__device__, AS1) address space; LDS destination is int4* in AS3.
// ---------------------------------------------------------------------------
#if defined(__has_builtin)
#if __has_builtin(__builtin_amdgcn_global_load_async_to_lds_b128)
#define HAVE_ASYNC_LDS 1
#endif
#endif
#ifndef HAVE_ASYNC_LDS
#define HAVE_ASYNC_LDS 0
#endif

__device__ __forceinline__ void async_b128(const bf16* g, bf16* l) {
#if HAVE_ASYNC_LDS
  v4i* gp = (v4i*)g;  // C-style cast drops const
  v4i* lp = (v4i*)l;
  __builtin_amdgcn_global_load_async_to_lds_b128((as1_v4i)gp, (as3_v4i)lp, 0, 0);
#else
  *(v8bf*)l = *(const v8bf*)g;
#endif
}

template <int N>
__device__ __forceinline__ void wait_async() {
#if HAVE_ASYNC_LDS
#if __has_builtin(__builtin_amdgcn_s_wait_asynccnt)
  __builtin_amdgcn_s_wait_asynccnt(N);
#else
  __asm__ volatile("s_wait_asynccnt %0" ::"i"(N) : "memory");
#endif
#endif
}

// ---------------------------------------------------------------------------
// WMMA helpers (CDNA5 wave32, 16x16x32 bf16 -> f32)
// ---------------------------------------------------------------------------
__device__ __forceinline__ v8f wmma_bf16(v16bf a, v16bf b, v8f c) {
  return __builtin_amdgcn_wmma_f32_16x16x32_bf16(false, a, false, b, (short)0, c,
                                                 false, false);
}

// A-matrix fragment (16x32, M x K): lane l holds row (l&15); element j ->
// K = k0 + ((l>>4)*16) + j
__device__ __forceinline__ v16bf load_a_frag(const bf16* base, int ld, int row,
                                             int k0, int lane) {
  const bf16* p = base + (size_t)row * ld + k0 + ((lane >> 4) << 4);
  v8bf lo = *(const v8bf*)(p);
  v8bf hi = *(const v8bf*)(p + 8);
  return __builtin_shufflevector(lo, hi, 0, 1, 2, 3, 4, 5, 6, 7,
                                 8, 9, 10, 11, 12, 13, 14, 15);
}

// B-matrix fragment (32x16, K x N) taken from a row-major [N][K] source:
// lane l covers column n0+(l&15); element j -> K = k0 + ((l>>4)*16) + j
__device__ __forceinline__ v16bf load_b_frag(const bf16* base, int ld, int n0,
                                             int k0, int lane) {
  const bf16* p = base + (size_t)(n0 + (lane & 15)) * ld + k0 + ((lane >> 4) << 4);
  return *(const v16bf*)(p);
}

// ---------------------------------------------------------------------------
// Blocked GEMM: C[M,N] = A[M,K] (bf16) @ B[N,K]^T (bf16) + bias[N]  (f32 out)
// 256 threads (8 waves) -> 64x128 C tile; LDS double-buffered 64-deep K
// stages filled with async global->LDS b128 copies (6 per thread per stage);
// each wave owns a 32x32 register tile (4 accumulators, 8 WMMA per stage).
// Requires M%64==0, N%128==0, K%64==0 (true for all uses here).
// ---------------------------------------------------------------------------
__global__ __launch_bounds__(256)
void gemm_bf16_wmma(const bf16* __restrict__ A, const bf16* __restrict__ B,
                    const float* __restrict__ bias, float* __restrict__ C,
                    int M, int N, int K) {
  __shared__ bf16 aS[2][64 * KSTEP];
  __shared__ bf16 bS[2][128 * KSTEP];
  int tid = threadIdx.x;
  int lane = tid & 31;
  int wave = tid >> 5;
  int n0 = blockIdx.x * 128;
  int m0 = blockIdx.y * 64;
  int wm = wave >> 2;  // 0..1
  int wn = wave & 3;   // 0..3

  v8f acc00 = {}, acc01 = {}, acc10 = {}, acc11 = {};

  auto issue = [&](int k, int buf) {
    // A panel: 64 rows x 64 elems = 512 x 16B chunks, 2 per thread
#pragma unroll
    for (int i = 0; i < 2; i++) {
      int chunk = tid * 2 + i;
      int row = chunk >> 3;
      int ce = (chunk & 7) * 8;
      async_b128(A + (size_t)(m0 + row) * K + k + ce, &aS[buf][row * KSTEP + ce]);
    }
    // B panel: 128 rows x 64 elems = 1024 x 16B chunks, 4 per thread
#pragma unroll
    for (int i = 0; i < 4; i++) {
      int chunk = tid * 4 + i;
      int row = chunk >> 3;
      int ce = (chunk & 7) * 8;
      async_b128(B + (size_t)(n0 + row) * K + k + ce, &bS[buf][row * KSTEP + ce]);
    }
  };

  issue(0, 0);
  int nk = K / KSTEP;
  for (int s = 0; s < nk; s++) {
    if (s + 1 < nk) {
      issue((s + 1) * KSTEP, (s + 1) & 1);
      wait_async<6>();  // drain stage s; stage s+1 (6 reqs) stays in flight
    } else {
      wait_async<0>();
    }
    __syncthreads();
    const bf16* aB = aS[s & 1];
    const bf16* bB = bS[s & 1];
#pragma unroll
    for (int ks = 0; ks < KSTEP; ks += 32) {
      v16bf a0 = load_a_frag(aB, KSTEP, wm * 32 + (lane & 15), ks, lane);
      v16bf a1 = load_a_frag(aB, KSTEP, wm * 32 + 16 + (lane & 15), ks, lane);
      v16bf b0 = load_b_frag(bB, KSTEP, wn * 32, ks, lane);
      v16bf b1 = load_b_frag(bB, KSTEP, wn * 32 + 16, ks, lane);
      acc00 = wmma_bf16(a0, b0, acc00);
      acc01 = wmma_bf16(a0, b1, acc01);
      acc10 = wmma_bf16(a1, b0, acc10);
      acc11 = wmma_bf16(a1, b1, acc11);
    }
    __syncthreads();
  }

  int mh = (lane >> 4) << 3;
  int nA = n0 + wn * 32 + (lane & 15);
  int nB = nA + 16;
  float bvA = bias ? bias[nA] : 0.f;
  float bvB = bias ? bias[nB] : 0.f;
#pragma unroll
  for (int r = 0; r < 8; r++) {
    int mA = m0 + wm * 32 + r + mh;
    int mB = mA + 16;
    C[(size_t)mA * N + nA] = acc00[r] + bvA;
    C[(size_t)mA * N + nB] = acc01[r] + bvB;
    C[(size_t)mB * N + nA] = acc10[r] + bvA;
    C[(size_t)mB * N + nB] = acc11[r] + bvB;
  }
}

// ---------------------------------------------------------------------------
// f32 -> bf16 convert (optional scale, used for DropConnect weight scaling)
// ---------------------------------------------------------------------------
__global__ void cvt_f32_bf16(const float* __restrict__ in, bf16* __restrict__ out,
                             long long n, float scale) {
  long long i = (long long)blockIdx.x * blockDim.x + threadIdx.x;
  if (i < n) out[i] = (bf16)(in[i] * scale);
}

// [B,S,H] f32 -> [B,NH,S,HD] bf16
__global__ void pack_heads(const float* __restrict__ in, bf16* __restrict__ out) {
  int i = blockIdx.x * 256 + threadIdx.x;
  if (i >= BB * SLEN * HDIM) return;
  int b = i / (SLEN * HDIM);
  int r = i - b * (SLEN * HDIM);
  int s = r / HDIM;
  int c = r - s * HDIM;
  int nh = c / HD;
  int d = c - nh * HD;
  out[(((size_t)(b * NHEAD + nh)) * SLEN + s) * HD + d] = (bf16)in[i];
}

// [B,S,H] f32 -> [B,NH,HD,S] bf16  (transposed pack for V: contiguous PV loads)
__global__ void pack_heads_t(const float* __restrict__ in, bf16* __restrict__ out) {
  int i = blockIdx.x * 256 + threadIdx.x;
  if (i >= BB * SLEN * HDIM) return;
  int b = i / (SLEN * HDIM);
  int r = i - b * (SLEN * HDIM);
  int s = r / HDIM;
  int c = r - s * HDIM;
  int nh = c / HD;
  int d = c - nh * HD;
  out[(((size_t)(b * NHEAD + nh)) * HD + d) * SLEN + s] = (bf16)in[i];
}

// ---------------------------------------------------------------------------
// PFN recurrent scan. Single persistent workgroup of 32 waves on one WGP.
// Dynamic LDS: gates f32 [16][3840] (245760B, aliased later by concat bf16
// [16][2304]) + h bf16 [16][768] (24576B) = 270336B (<320KB WGP LDS).
// ---------------------------------------------------------------------------
__global__ __launch_bounds__(1024)
void pfn_scan(const float* __restrict__ xg, const bf16* __restrict__ Whb,
              const float* __restrict__ bh, const bf16* __restrict__ Wtb,
              const float* __restrict__ bt, float* __restrict__ cbuf,
              float* __restrict__ h0f, float* __restrict__ h1f,
              bf16* __restrict__ h0b, bf16* __restrict__ h1b) {
  extern __shared__ char smem_raw[];
  float* gates = (float*)smem_raw;                                 // [16][3840]
  bf16* concatb = (bf16*)smem_raw;                                 // [16][2304] aliased
  bf16* hls = (bf16*)(smem_raw + (size_t)BB * H5 * sizeof(float)); // [16][768]

  int tid = threadIdx.x;
  int lane = tid & 31;
  int wave = tid >> 5;

  for (int i = tid; i < BB * HDIM; i += 1024) {
    hls[i] = (bf16)0.0f;
    cbuf[i] = 0.0f;
  }
  __syncthreads();

#pragma unroll 1
  for (int t = 0; t < SLEN; ++t) {
    // ---- Phase A: gates = xg[t] + h @ Wh_eff^T + bh  (M=16, N=3840, K=768)
    for (int nt = wave; nt < H5 / 16; nt += 32) {
      int n0 = nt * 16;
      v8f acc = {};
#pragma unroll 4
      for (int k = 0; k < HDIM; k += 32) {
        v16bf a = load_a_frag(hls, HDIM, lane & 15, k, lane);
        v16bf b = load_b_frag(Whb, HDIM, n0, k, lane);
        acc = wmma_bf16(a, b, acc);
      }
      int n = n0 + (lane & 15);
      float bv = bh[n];
      int mh = (lane >> 4) << 3;
#pragma unroll
      for (int r = 0; r < 8; r++) {
        int m = r + mh;
        gates[m * H5 + n] = acc[r] + xg[(size_t)(t * BB + m) * H5 + n] + bv;
      }
    }
    __syncthreads();

    // ---- Phase B: cumsoftmax over 4 gate groups x 16 batch rows (2 per wave)
    for (int rr = 0; rr < 2; rr++) {
      int rowid = wave * 2 + rr;  // 0..63
      int g = rowid >> 4;         // group (eg_cin, rg_cin, eg_c, rg_c)
      int b = rowid & 15;
      float* rowp = gates + b * H5 + (g + 1) * HDIM;
      float vals[24];
      float mx = -1e30f;
#pragma unroll
      for (int j = 0; j < 24; j++) {
        vals[j] = rowp[lane * 24 + j];
        mx = fmaxf(mx, vals[j]);
      }
      for (int d = 1; d < 32; d <<= 1) mx = fmaxf(mx, __shfl_xor(mx, d, 32));
      float run = 0.f;
#pragma unroll
      for (int j = 0; j < 24; j++) {
        run += __expf(vals[j] - mx);
        vals[j] = run;  // inclusive in-lane cumsum of exp
      }
      float inc = run;  // wave-wide inclusive scan of lane totals
      for (int d = 1; d < 32; d <<= 1) {
        float tt = __shfl_up(inc, d, 32);
        if (lane >= d) inc += tt;
      }
      float total = __shfl(inc, 31, 32);
      float excl = inc - run;
      float invt = 1.0f / total;
      bool flip = (g == 0) || (g == 2);  // erase gates use 1 - cumsoftmax
#pragma unroll
      for (int j = 0; j < 24; j++) {
        float cs = (excl + vals[j]) * invt;
        rowp[lane * 24 + j] = flip ? (1.0f - cs) : cs;
      }
    }
    __syncthreads();

    // ---- Phase C: elementwise gate mixing (read gates, barrier, then
    // overwrite the same LDS with concat bf16 = [c_re | c_ner | share])
    float cre[12], cner[12], csh[12];
#pragma unroll
    for (int j = 0; j < 12; j++) {
      int idx = tid + 1024 * j;  // 0..12287 -> (b,k)
      int b = idx / HDIM;
      int k = idx - b * HDIM;
      float cv = tanhf(gates[b * H5 + k]);
      float eg_cin = gates[b * H5 + HDIM + k];
      float rg_cin = gates[b * H5 + 2 * HDIM + k];
      float eg_c = gates[b * H5 + 3 * HDIM + k];
      float rg_c = gates[b * H5 + 4 * HDIM + k];
      float cin = cbuf[idx];
      float ov_c = rg_c * eg_c;
      float up_c = rg_c - ov_c;
      float dn_c = eg_c - ov_c;
      float ov_i = rg_cin * eg_cin;
      float up_i = rg_cin - ov_i;
      float dn_i = eg_cin - ov_i;
      float share = ov_i * cin + ov_c * cv;
      float c_re = up_i * cin + up_c * cv + share;
      float c_ner = dn_i * cin + dn_c * cv + share;
      cre[j] = c_re;
      cner[j] = c_ner;
      csh[j] = share;
      float h_re = tanhf(c_re);
      float h_ner = tanhf(c_ner);
      size_t go = ((size_t)b * SLEN + t) * HDIM + k;  // [B,S,H] layout
      h0f[go] = h_ner;
      h1f[go] = h_re;
      h0b[go] = (bf16)h_ner;
      h1b[go] = (bf16)h_re;
    }
    __syncthreads();
#pragma unroll
    for (int j = 0; j < 12; j++) {
      int idx = tid + 1024 * j;
      int b = idx / HDIM;
      int k = idx - b * HDIM;
      concatb[b * H3 + k] = (bf16)cre[j];
      concatb[b * H3 + HDIM + k] = (bf16)cner[j];
      concatb[b * H3 + 2 * HDIM + k] = (bf16)csh[j];
    }
    __syncthreads();

    // ---- Phase D: c_out = concat @ Wt^T + bt ; h = tanh(c_out)
    for (int nt = wave; nt < HDIM / 16; nt += 32) {
      int n0 = nt * 16;
      v8f acc = {};
#pragma unroll 4
      for (int k = 0; k < H3; k += 32) {
        v16bf a = load_a_frag(concatb, H3, lane & 15, k, lane);
        v16bf b = load_b_frag(Wtb, H3, n0, k, lane);
        acc = wmma_bf16(a, b, acc);
      }
      int n = n0 + (lane & 15);
      float bv = bt[n];
      int mh = (lane >> 4) << 3;
#pragma unroll
      for (int r = 0; r < 8; r++) {
        int m = r + mh;
        float co = acc[r] + bv;
        cbuf[m * HDIM + n] = co;
        hls[m * HDIM + n] = (bf16)tanhf(co);
      }
    }
    __threadfence_block();
    __syncthreads();
  }
}

// ---------------------------------------------------------------------------
// Flash-style attention with relative position bias.
// grid = B*NH*(S/128); block = 256 (8 waves); wave owns 16 query rows.
// scores = Q K^T + Q RP^T (reference has no 1/sqrt(d)); online softmax; O=P V.
// V is pre-transposed per head ([B,NH,HD,S]) so PV B-fragments are contiguous.
// ---------------------------------------------------------------------------
__global__ __launch_bounds__(256)
void flash_attn(const bf16* __restrict__ qh, const bf16* __restrict__ kh,
                const bf16* __restrict__ vt, const bf16* __restrict__ rp,
                float* __restrict__ o) {
  __shared__ bf16 p_smem[8][16][32];
  int lane = threadIdx.x & 31;
  int wave = threadIdx.x >> 5;
  int bh = blockIdx.x >> 2;  // b*NH + n
  int qblk = blockIdx.x & 3;
  int b = bh / NHEAD;
  int n = bh - b * NHEAD;
  int qrow0 = qblk * 128 + wave * 16;

  const bf16* qbase = qh + (size_t)bh * SLEN * HD;
  const bf16* kbase = kh + (size_t)bh * SLEN * HD;
  const bf16* vbase = vt + (size_t)bh * SLEN * HD;  // [HD][S]
  const bf16* rbase = rp + (size_t)n * SLEN * HD;

  int rowq = qrow0 + (lane & 15);
  v16bf aq0 = load_a_frag(qbase, HD, rowq, 0, lane);
  v16bf aq1 = load_a_frag(qbase, HD, rowq, 32, lane);

  float m_r[8], l_r[8];
  v8f oacc[4];
  v8f vzero = {};
#pragma unroll
  for (int r = 0; r < 8; r++) {
    m_r[r] = -1e30f;
    l_r[r] = 0.f;
  }
#pragma unroll
  for (int dc = 0; dc < 4; dc++) oacc[dc] = vzero;

  for (int kt = 0; kt < SLEN; kt += 32) {
    v8f s0 = {}, s1 = {};
    {
      v16bf bk;
      bk = load_b_frag(kbase, HD, kt, 0, lane);       s0 = wmma_bf16(aq0, bk, s0);
      bk = load_b_frag(kbase, HD, kt, 32, lane);      s0 = wmma_bf16(aq1, bk, s0);
      bk = load_b_frag(rbase, HD, kt, 0, lane);       s0 = wmma_bf16(aq0, bk, s0);
      bk = load_b_frag(rbase, HD, kt, 32, lane);      s0 = wmma_bf16(aq1, bk, s0);
      bk = load_b_frag(kbase, HD, kt + 16, 0, lane);  s1 = wmma_bf16(aq0, bk, s1);
      bk = load_b_frag(kbase, HD, kt + 16, 32, lane); s1 = wmma_bf16(aq1, bk, s1);
      bk = load_b_frag(rbase, HD, kt + 16, 0, lane);  s1 = wmma_bf16(aq0, bk, s1);
      bk = load_b_frag(rbase, HD, kt + 16, 32, lane); s1 = wmma_bf16(aq1, bk, s1);
    }
    // online softmax: row m = r + 8*(lane>>4) lives in 16 lanes of a half
#pragma unroll
    for (int r = 0; r < 8; r++) {
      float mx = fmaxf(s0[r], s1[r]);
      for (int d = 1; d < 16; d <<= 1) mx = fmaxf(mx, __shfl_xor(mx, d, 32));
      float mnew = fmaxf(m_r[r], mx);
      float scale = __expf(m_r[r] - mnew);
      float p0 = __expf(s0[r] - mnew);
      float p1 = __expf(s1[r] - mnew);
      float rs = p0 + p1;
      for (int d = 1; d < 16; d <<= 1) rs += __shfl_xor(rs, d, 32);
      l_r[r] = l_r[r] * scale + rs;
      m_r[r] = mnew;
#pragma unroll
      for (int dc = 0; dc < 4; dc++) oacc[dc][r] *= scale;
      s0[r] = p0;
      s1[r] = p1;
    }
    // stage P (16x32) through LDS to re-fragment as WMMA A operand
    int mh = (lane >> 4) << 3;
#pragma unroll
    for (int r = 0; r < 8; r++) {
      p_smem[wave][r + mh][lane & 15] = (bf16)s0[r];
      p_smem[wave][r + mh][16 + (lane & 15)] = (bf16)s1[r];
    }
    __asm__ volatile("s_wait_dscnt 0" ::: "memory");  // wave-local LDS turnaround
    v16bf ap = load_a_frag(&p_smem[wave][0][0], 32, lane & 15, 0, lane);
#pragma unroll
    for (int dc = 0; dc < 4; dc++) {
      v16bf bv = load_b_frag(vbase, SLEN, dc * 16, kt, lane);  // V^T: contiguous
      oacc[dc] = wmma_bf16(ap, bv, oacc[dc]);
    }
  }
  // epilogue: normalize and write O into [B,S,H] f32
  int mh = (lane >> 4) << 3;
#pragma unroll
  for (int r = 0; r < 8; r++) {
    float inv = 1.0f / l_r[r];
    int m = qrow0 + r + mh;
#pragma unroll
    for (int dc = 0; dc < 4; dc++) {
      o[((size_t)(b * SLEN + m)) * HDIM + n * HD + dc * 16 + (lane & 15)] =
          oacc[dc][r] * inv;
    }
  }
}

// ---------------------------------------------------------------------------
// out[s,b,:] = LayerNorm(h[b,s,:] + proj[b,s,:]) * g + beta
// block = 256 (8 waves), one 768-row per wave.
// ---------------------------------------------------------------------------
__global__ __launch_bounds__(256)
void residual_ln(const float* __restrict__ h, const float* __restrict__ proj,
                 const float* __restrict__ g, const float* __restrict__ be,
                 float* __restrict__ out) {
  int lane = threadIdx.x & 31;
  int wave = threadIdx.x >> 5;
  int row = blockIdx.x * 8 + wave;  // b*S + s
  int b = row / SLEN;
  int s = row - b * SLEN;
  const float* hp = h + (size_t)row * HDIM;
  const float* pp = proj + (size_t)row * HDIM;
  float v[24];
  float sum = 0.f, sq = 0.f;
#pragma unroll
  for (int j = 0; j < 24; j++) {
    int e = lane + 32 * j;
    float x = hp[e] + pp[e];
    v[j] = x;
    sum += x;
    sq += x * x;
  }
  for (int d = 1; d < 32; d <<= 1) {
    sum += __shfl_xor(sum, d, 32);
    sq += __shfl_xor(sq, d, 32);
  }
  float mean = sum * (1.0f / HDIM);
  float var = sq * (1.0f / HDIM) - mean * mean;
  float rstd = rsqrtf(var + 1e-5f);
  float* op = out + ((size_t)s * BB + b) * HDIM;
#pragma unroll
  for (int j = 0; j < 24; j++) {
    int e = lane + 32 * j;
    op[e] = (v[j] - mean) * rstd * g[e] + be[e];
  }
}

// ---------------------------------------------------------------------------
// Host launcher
// ---------------------------------------------------------------------------
extern "C" void kernel_launch(void* const* d_in, const int* in_sizes, int n_in,
                              void* d_out, int out_size, void* d_ws, size_t ws_size,
                              hipStream_t stream) {
  (void)in_sizes; (void)n_in; (void)out_size; (void)ws_size;
  const float* x     = (const float*)d_in[0];
  const float* Wi    = (const float*)d_in[1];
  const float* bi    = (const float*)d_in[2];
  const float* Wh    = (const float*)d_in[3];
  const float* bh    = (const float*)d_in[4];
  const float* Wt    = (const float*)d_in[5];
  const float* bt    = (const float*)d_in[6];
  const float* ln_g  = (const float*)d_in[7];
  const float* ln_b  = (const float*)d_in[8];
  const float* qW    = (const float*)d_in[9];
  const float* qb    = (const float*)d_in[10];
  const float* kW    = (const float*)d_in[11];
  const float* kb    = (const float*)d_in[12];
  const float* vW    = (const float*)d_in[13];
  const float* vb    = (const float*)d_in[14];
  const float* oW    = (const float*)d_in[15];
  const float* obias = (const float*)d_in[16];
  const float* relp  = (const float*)d_in[17];

  char* ws = (char*)d_ws;
  size_t off = 0;
  auto alloc = [&](size_t bytes) -> void* {
    void* p = ws + off;
    off += (bytes + 255) & ~(size_t)255;
    return p;
  };

  const size_t MBS = (size_t)SLEN * BB;  // 8192 rows
  bf16* x_b  = (bf16*)alloc(MBS * HDIM * 2);
  bf16* Wi_b = (bf16*)alloc((size_t)H5 * HDIM * 2);
  bf16* Wh_b = (bf16*)alloc((size_t)H5 * HDIM * 2);
  bf16* Wt_b = (bf16*)alloc((size_t)HDIM * H3 * 2);
  bf16* qW_b = (bf16*)alloc((size_t)3 * HDIM * HDIM * 2);
  bf16* kW_b = (bf16*)alloc((size_t)3 * HDIM * HDIM * 2);
  bf16* vW_b = (bf16*)alloc((size_t)3 * HDIM * HDIM * 2);
  bf16* oW_b = (bf16*)alloc((size_t)3 * HDIM * HDIM * 2);
  bf16* rp_b = (bf16*)alloc((size_t)3 * NHEAD * SLEN * HD * 2);
  float* xg  = (float*)alloc(MBS * H5 * 4);
  float* cbuf = (float*)alloc((size_t)BB * HDIM * 4);
  float* h0f = (float*)alloc(MBS * HDIM * 4);
  float* h1f = (float*)alloc(MBS * HDIM * 4);
  bf16* h0b = (bf16*)alloc(MBS * HDIM * 2);
  bf16* h1b = (bf16*)alloc(MBS * HDIM * 2);
  float* qf = (float*)alloc(MBS * HDIM * 4);
  float* kf = (float*)alloc(MBS * HDIM * 4);
  float* vf = (float*)alloc(MBS * HDIM * 4);
  bf16* qhb = (bf16*)alloc(MBS * HDIM * 2);
  bf16* khb = (bf16*)alloc(MBS * HDIM * 2);
  bf16* vtb = (bf16*)alloc(MBS * HDIM * 2);
  float* of = (float*)alloc(MBS * HDIM * 4);
  float* pj = (float*)alloc(MBS * HDIM * 4);
  bf16* ofb = (bf16*)alloc(MBS * HDIM * 2);

  auto cvt = [&](const float* in, bf16* o_, size_t n, float sc) {
    cvt_f32_bf16<<<dim3((unsigned)((n + 255) / 256)), dim3(256), 0, stream>>>(
        in, o_, (long long)n, sc);
  };
  cvt(x, x_b, MBS * HDIM, 1.f);
  cvt(Wi, Wi_b, (size_t)H5 * HDIM, 1.f);
  cvt(Wh, Wh_b, (size_t)H5 * HDIM, 0.9f);  // DropConnect eval scaling (1-p)
  cvt(Wt, Wt_b, (size_t)HDIM * H3, 1.f);
  cvt(qW, qW_b, (size_t)3 * HDIM * HDIM, 1.f);
  cvt(kW, kW_b, (size_t)3 * HDIM * HDIM, 1.f);
  cvt(vW, vW_b, (size_t)3 * HDIM * HDIM, 1.f);
  cvt(oW, oW_b, (size_t)3 * HDIM * HDIM, 1.f);
  cvt(relp, rp_b, (size_t)3 * NHEAD * SLEN * HD, 1.f);

  // x_gates = x @ Wi^T + bi   [8192 x 3840]
  gemm_bf16_wmma<<<dim3(H5 / 128, (unsigned)(MBS / 64)), 256, 0, stream>>>(
      x_b, Wi_b, bi, xg, (int)MBS, H5, HDIM);

  // recurrent scan (single persistent workgroup, 270336B dynamic LDS)
  const size_t scan_lds = (size_t)BB * H5 * sizeof(float) + (size_t)BB * HDIM * 2;
  pfn_scan<<<1, 1024, (unsigned)scan_lds, stream>>>(xg, Wh_b, bh, Wt_b, bt, cbuf,
                                                    h0f, h1f, h0b, h1b);

  const unsigned pack_grid = (unsigned)((MBS * HDIM + 255) / 256);
  for (int lay = 0; lay < 3; lay++) {
    // faithful reference bug: layers 1 and 2 both attend over h_re
    const bf16* hb = (lay == 0) ? h0b : h1b;
    const float* hf = (lay == 0) ? h0f : h1f;
    gemm_bf16_wmma<<<dim3(HDIM / 128, (unsigned)(MBS / 64)), 256, 0, stream>>>(
        hb, qW_b + (size_t)lay * HDIM * HDIM, qb + lay * HDIM, qf, (int)MBS, HDIM,
        HDIM);
    gemm_bf16_wmma<<<dim3(HDIM / 128, (unsigned)(MBS / 64)), 256, 0, stream>>>(
        hb, kW_b + (size_t)lay * HDIM * HDIM, kb + lay * HDIM, kf, (int)MBS, HDIM,
        HDIM);
    gemm_bf16_wmma<<<dim3(HDIM / 128, (unsigned)(MBS / 64)), 256, 0, stream>>>(
        hb, vW_b + (size_t)lay * HDIM * HDIM, vb + lay * HDIM, vf, (int)MBS, HDIM,
        HDIM);
    pack_heads<<<pack_grid, 256, 0, stream>>>(qf, qhb);
    pack_heads<<<pack_grid, 256, 0, stream>>>(kf, khb);
    pack_heads_t<<<pack_grid, 256, 0, stream>>>(vf, vtb);
    flash_attn<<<dim3(BB * NHEAD * (SLEN / 128)), 256, 0, stream>>>(
        qhb, khb, vtb, rp_b + (size_t)lay * NHEAD * SLEN * HD, of);
    cvt(of, ofb, MBS * HDIM, 1.f);
    gemm_bf16_wmma<<<dim3(HDIM / 128, (unsigned)(MBS / 64)), 256, 0, stream>>>(
        ofb, oW_b + (size_t)lay * HDIM * HDIM, obias + lay * HDIM, pj, (int)MBS,
        HDIM, HDIM);
    residual_ln<<<dim3((unsigned)(MBS / 8)), 256, 0, stream>>>(
        hf, pj, ln_g, ln_b, (float*)d_out + (size_t)lay * MBS * HDIM);
  }
}
